// MultiHeadSelfAttention_88587995448047
// MI455X (gfx1250) — compile-verified, matrix operand order checked by reading, not run on hardware
//
#include <hip/hip_runtime.h>
#include <hip/hip_bf16.h>
#include <stdint.h>

#define H_NUM 16
#define DH 64
#define SEQ 2048
#define BATCH 2
#define DMODEL 1024
#define NROWS (BATCH*SEQ)   // 4096

#if __has_builtin(__builtin_amdgcn_global_load_async_to_lds_b128) && \
    __has_builtin(__builtin_amdgcn_s_wait_asynccnt)
#define ASYNC_OK 1
#else
#define ASYNC_OK 0
#endif

#if defined(__AMDGCN__) && !ASYNC_OK
#warning "gfx1250 async-to-LDS builtins NOT available; falling back to direct global bias loads"
#endif

typedef __bf16 v16bf __attribute__((ext_vector_type(16)));
typedef float  v8f   __attribute__((ext_vector_type(8)));
typedef int    v4i   __attribute__((vector_size(16)));   // matches builtin param type

union FragBF {
    v16bf v;
    uint4 q[2];
};

__device__ __forceinline__ unsigned short f32_to_bf16(float f) {
    union { float f; unsigned int u; } c; c.f = f;
    unsigned int u = c.u;
    unsigned int r = u + 0x7FFFu + ((u >> 16) & 1u);   // round-to-nearest-even
    return (unsigned short)(r >> 16);
}

__global__ void cvt_f32_bf16(const float* __restrict__ in,
                             unsigned short* __restrict__ out, int n) {
    int i = blockIdx.x * blockDim.x + threadIdx.x;
    int stride = gridDim.x * blockDim.x;
    for (; i < n; i += stride) out[i] = f32_to_bf16(in[i]);
}

// y = X @ W^T + b ; X:[NROWS,DMODEL] bf16, W:[DMODEL(out),DMODEL(in)] bf16
// out: transposed==0 -> [B,H,S,DH] ; transposed==1 -> [B,H,DH,S]
// Software-pipelined: fragments for k0+32 are loaded while WMMAs for k0 issue.
__global__ __launch_bounds__(128)
void proj_gemm(const unsigned short* __restrict__ X,
               const unsigned short* __restrict__ W,
               const float* __restrict__ bias,
               unsigned short* __restrict__ out,
               int transposed)
{
    const int lane = threadIdx.x & 31;
    const int wave = threadIdx.x >> 5;
    const int row0 = blockIdx.x * 64 + wave * 16;
    const int col0 = blockIdx.y * 64;
    const int lrow = lane & 15;
    const int half = lane >> 4;

    v8f zero = {};
    v8f acc[4];
    #pragma unroll
    for (int t = 0; t < 4; ++t) acc[t] = zero;

    const unsigned short* arow = X + (size_t)(row0 + lrow) * DMODEL + 8 * half;
    const unsigned short* brow[4];
    #pragma unroll
    for (int t = 0; t < 4; ++t)
        brow[t] = W + (size_t)(col0 + t * 16 + lrow) * DMODEL + 8 * half;

    FragBF a_cur, b_cur[4];
    a_cur.q[0] = *(const uint4*)(arow);
    a_cur.q[1] = *(const uint4*)(arow + 16);
    #pragma unroll
    for (int t = 0; t < 4; ++t) {
        b_cur[t].q[0] = *(const uint4*)(brow[t]);
        b_cur[t].q[1] = *(const uint4*)(brow[t] + 16);
    }

    for (int k0 = 0; k0 < DMODEL; k0 += 32) {
        FragBF a_nxt, b_nxt[4];
        const bool more = (k0 + 32) < DMODEL;
        if (more) {
            const unsigned short* ap = arow + k0 + 32;
            a_nxt.q[0] = *(const uint4*)(ap);
            a_nxt.q[1] = *(const uint4*)(ap + 16);
            #pragma unroll
            for (int t = 0; t < 4; ++t) {
                const unsigned short* bp = brow[t] + k0 + 32;
                b_nxt[t].q[0] = *(const uint4*)(bp);
                b_nxt[t].q[1] = *(const uint4*)(bp + 16);
            }
        }
        #pragma unroll
        for (int t = 0; t < 4; ++t)
            acc[t] = __builtin_amdgcn_wmma_f32_16x16x32_bf16(
                false, a_cur.v, false, b_cur[t].v, (short)0, acc[t], false, false);
        if (more) {
            a_cur = a_nxt;
            #pragma unroll
            for (int t = 0; t < 4; ++t) b_cur[t] = b_nxt[t];
        }
    }

    #pragma unroll
    for (int t = 0; t < 4; ++t) {
        int ocol = col0 + t * 16 + lrow;       // output feature
        int h = ocol >> 6, d = ocol & 63;
        float bv = bias[ocol];
        #pragma unroll
        for (int g = 0; g < 8; ++g) {
            int n = row0 + g + 8 * half;       // global row
            int b = n >> 11, s = n & (SEQ - 1);
            unsigned short val = f32_to_bf16(acc[t][g] + bv);
            size_t idx;
            if (transposed) idx = ((size_t)(b * H_NUM + h) * DH + d) * SEQ + s;
            else            idx = ((size_t)(b * H_NUM + h) * SEQ + s) * DH + d;
            out[idx] = val;
        }
    }
}

#if ASYNC_OK
// Async-copy one 64x64 f32 bias tile (16KB) into LDS: 1024 b128 transfers,
// 8 per thread (= 8 ASYNCcnt increments per wave).
__device__ __forceinline__ void async_copy_bias_tile(const float* gtile,  // &bias[q0blk*SEQ + kt]
                                                     float* ltile,       // LDS dest (64*64 floats)
                                                     int tid)
{
    #pragma unroll
    for (int i = 0; i < 8; ++i) {
        int j = tid + 128 * i;        // 0..1023
        int r = j >> 4;               // row 0..63
        int c = (j & 15) << 2;        // col (floats)
        const float* g = gtile + (size_t)r * SEQ + c;
        float* l = ltile + r * 64 + c;
        __builtin_amdgcn_global_load_async_to_lds_b128(
            (__attribute__((address_space(1))) v4i*)g,
            (__attribute__((address_space(3))) v4i*)l,
            0, 0);
    }
}
#endif

// Flash attention: Q,K [B,H,S,DH] bf16; Vt [B,H,DH,S] bf16; bias [1,H,S,S] f32;
// mask [B,S] i32; out [B,S,D] f32
__global__ __launch_bounds__(128)
void attn_kernel(const unsigned short* __restrict__ Q,
                 const unsigned short* __restrict__ K,
                 const unsigned short* __restrict__ Vt,
                 const float* __restrict__ bias,
                 const int* __restrict__ mask,
                 float* __restrict__ out)
{
    __shared__ __align__(16) unsigned short Plds[4 * 16 * 64];
#if ASYNC_OK
    __shared__ __align__(16) float BiasLds[2 * 64 * 64];
#endif
    const int tid = threadIdx.x;
    const int lane = tid & 31;
    const int wave = tid >> 5;
    const int bh = blockIdx.x;                 // 0..B*H-1
    const int b = bh >> 4, h = bh & 15;
    const int q0blk = blockIdx.y * 64;
    const int q0 = q0blk + wave * 16;
    const int lrow = lane & 15;
    const int half = lane >> 4;

    const unsigned short* Qbase = Q + (size_t)bh * SEQ * DH;
    const unsigned short* Kbase = K + (size_t)bh * SEQ * DH;
    const unsigned short* Vbase = Vt + (size_t)bh * DH * SEQ;
    const float* biasbase = bias + (size_t)h * SEQ * SEQ;
    const int* maskbase = mask + b * SEQ;

    // Q fragments: features [0,32) and [32,64)
    FragBF qa0, qa1;
    {
        const unsigned short* qp = Qbase + (size_t)(q0 + lrow) * DH + 8 * half;
        qa0.q[0] = *(const uint4*)(qp);
        qa0.q[1] = *(const uint4*)(qp + 16);
        qa1.q[0] = *(const uint4*)(qp + 32);
        qa1.q[1] = *(const uint4*)(qp + 48);
    }

    v8f zero = {};
    v8f ctx[4];
    #pragma unroll
    for (int t = 0; t < 4; ++t) ctx[t] = zero;
    float mrow[8], lsum[8];
    #pragma unroll
    for (int g = 0; g < 8; ++g) { mrow[g] = -__builtin_inff(); lsum[g] = 0.f; }

    unsigned short* Pw = Plds + wave * 16 * 64;

#if ASYNC_OK
    // prologue: stage bias tile for kt=0 into buffer 0
    async_copy_bias_tile(biasbase + (size_t)q0blk * SEQ, BiasLds, tid);
    int cur = 0;
#endif

    for (int kt = 0; kt < SEQ; kt += 64) {
        const bool has_next = (kt + 64) < SEQ;
#if ASYNC_OK
        // stage NEXT bias tile while this iteration computes
        if (has_next)
            async_copy_bias_tile(biasbase + (size_t)q0blk * SEQ + kt + 64,
                                 BiasLds + (cur ^ 1) * (64 * 64), tid);
#endif
        // ---- scores: Q(16x64) x K^T(64x64) as 4 tiles of 2 WMMAs ----
        v8f s[4];
        #pragma unroll
        for (int t = 0; t < 4; ++t) {
            FragBF kb0, kb1;
            const unsigned short* kp = Kbase + (size_t)(kt + t * 16 + lrow) * DH + 8 * half;
            kb0.q[0] = *(const uint4*)(kp);
            kb0.q[1] = *(const uint4*)(kp + 16);
            kb1.q[0] = *(const uint4*)(kp + 32);
            kb1.q[1] = *(const uint4*)(kp + 48);
            v8f a = __builtin_amdgcn_wmma_f32_16x16x32_bf16(
                false, qa0.v, false, kb0.v, (short)0, zero, false, false);
            a = __builtin_amdgcn_wmma_f32_16x16x32_bf16(
                false, qa1.v, false, kb1.v, (short)0, a, false, false);
            s[t] = a;
        }
        // ---- bias (pre-scale, as in reference), 1/sqrt(64), additive mask ----
#if ASYNC_OK
        // current tile was staged by ALL waves: wait own ASYNCcnt, then barrier.
        if (has_next) __builtin_amdgcn_s_wait_asynccnt(8);
        else          __builtin_amdgcn_s_wait_asynccnt(0);
        __syncthreads();
        const float* Bt = BiasLds + cur * (64 * 64);
#endif
        #pragma unroll
        for (int t = 0; t < 4; ++t) {
            int kcol = kt + t * 16 + lrow;
            float madd = (1.0f - (float)maskbase[kcol]) * -3.0e38f;
            #pragma unroll
            for (int g = 0; g < 8; ++g) {
#if ASYNC_OK
                float bval = Bt[(wave * 16 + g + 8 * half) * 64 + t * 16 + lrow];
#else
                int qr = q0 + g + 8 * half;
                float bval = biasbase[(size_t)qr * SEQ + kcol];
#endif
                s[t][g] = (s[t][g] + bval) * 0.125f + madd;
            }
        }
        // ---- online softmax (row = vgpr g + 8*half; reduce over 16-lane half) ----
        float mnew[8], scl[8];
        #pragma unroll
        for (int g = 0; g < 8; ++g) {
            float mt = fmaxf(fmaxf(s[0][g], s[1][g]), fmaxf(s[2][g], s[3][g]));
            #pragma unroll
            for (int off = 1; off < 16; off <<= 1)
                mt = fmaxf(mt, __shfl_xor(mt, off, 32));
            float mn = fmaxf(mrow[g], mt);
            scl[g] = __expf(mrow[g] - mn);
            mnew[g] = mn;
            mrow[g] = mn;
        }
        #pragma unroll
        for (int g = 0; g < 8; ++g) {
            float rs = 0.f;
            #pragma unroll
            for (int t = 0; t < 4; ++t) {
                float p = __expf(s[t][g] - mnew[g]);
                s[t][g] = p;
                rs += p;
            }
            #pragma unroll
            for (int off = 1; off < 16; off <<= 1)
                rs += __shfl_xor(rs, off, 32);
            lsum[g] = lsum[g] * scl[g] + rs;
            #pragma unroll
            for (int t = 0; t < 4; ++t) ctx[t][g] *= scl[g];
        }
        // ---- P (C/D layout) -> wave-private LDS -> A-fragment layout ----
        // (LDS is in-order per wave; region is wave-private, no barrier needed)
        #pragma unroll
        for (int t = 0; t < 4; ++t) {
            int col = t * 16 + lrow;
            #pragma unroll
            for (int g = 0; g < 8; ++g)
                Pw[(g + 8 * half) * 64 + col] = f32_to_bf16(s[t][g]);
        }
        FragBF pa0, pa1;
        {
            const unsigned short* pp = Pw + lrow * 64 + 8 * half;
            pa0.q[0] = *(const uint4*)(pp);
            pa0.q[1] = *(const uint4*)(pp + 16);
            pa1.q[0] = *(const uint4*)(pp + 32);
            pa1.q[1] = *(const uint4*)(pp + 48);
        }
        // ---- ctx += P(16x64) x V(64x64); V transposed so B-frags are b128 ----
        #pragma unroll
        for (int t = 0; t < 4; ++t) {
            FragBF vb0, vb1;
            const unsigned short* vp = Vbase + (size_t)(t * 16 + lrow) * SEQ + kt + 8 * half;
            vb0.q[0] = *(const uint4*)(vp);
            vb0.q[1] = *(const uint4*)(vp + 16);
            vb1.q[0] = *(const uint4*)(vp + 32);
            vb1.q[1] = *(const uint4*)(vp + 48);
            ctx[t] = __builtin_amdgcn_wmma_f32_16x16x32_bf16(
                false, pa0.v, false, vb0.v, (short)0, ctx[t], false, false);
            ctx[t] = __builtin_amdgcn_wmma_f32_16x16x32_bf16(
                false, pa1.v, false, vb1.v, (short)0, ctx[t], false, false);
        }
        // end-of-iteration barrier: all waves done reading bias buffer `cur`
        // before it is overwritten by the async stage in the next iteration.
        __syncthreads();
#if ASYNC_OK
        cur ^= 1;
#endif
    }

    // ---- epilogue: ctx / l, fp32 out [B,S,D] ----
    #pragma unroll
    for (int t = 0; t < 4; ++t) {
        int d = t * 16 + lrow;
        #pragma unroll
        for (int g = 0; g < 8; ++g) {
            int qr = q0 + g + 8 * half;
            out[((size_t)b * SEQ + qr) * DMODEL + h * DH + d] = ctx[t][g] / lsum[g];
        }
    }
}

extern "C" void kernel_launch(void* const* d_in, const int* in_sizes, int n_in,
                              void* d_out, int out_size, void* d_ws, size_t ws_size,
                              hipStream_t stream)
{
    const float* hidden = (const float*)d_in[0];
    const int*   mask   = (const int*)d_in[1];
    const float* bias   = (const float*)d_in[2];
    const float* Wq     = (const float*)d_in[3];
    const float* bq     = (const float*)d_in[4];
    const float* Wk     = (const float*)d_in[5];
    const float* bk     = (const float*)d_in[6];
    const float* Wv     = (const float*)d_in[7];
    const float* bv     = (const float*)d_in[8];
    float* out = (float*)d_out;

    unsigned short* Xb  = (unsigned short*)d_ws;                  // [4096,1024]
    unsigned short* Wqb = Xb  + (size_t)NROWS * DMODEL;
    unsigned short* Wkb = Wqb + (size_t)DMODEL * DMODEL;
    unsigned short* Wvb = Wkb + (size_t)DMODEL * DMODEL;
    unsigned short* Qb  = Wvb + (size_t)DMODEL * DMODEL;          // [B,H,S,DH]
    unsigned short* Kb  = Qb  + (size_t)NROWS * DMODEL;           // [B,H,S,DH]
    unsigned short* Vtb = Kb  + (size_t)NROWS * DMODEL;           // [B,H,DH,S]

    const int nX = NROWS * DMODEL;
    const int nW = DMODEL * DMODEL;
    cvt_f32_bf16<<<(nX + 255) / 256, 256, 0, stream>>>(hidden, Xb, nX);
    cvt_f32_bf16<<<(nW + 255) / 256, 256, 0, stream>>>(Wq, Wqb, nW);
    cvt_f32_bf16<<<(nW + 255) / 256, 256, 0, stream>>>(Wk, Wkb, nW);
    cvt_f32_bf16<<<(nW + 255) / 256, 256, 0, stream>>>(Wv, Wvb, nW);

    dim3 pg(NROWS / 64, DMODEL / 64);
    proj_gemm<<<pg, 128, 0, stream>>>(Xb, Wqb, bq, Qb, 0);
    proj_gemm<<<pg, 128, 0, stream>>>(Xb, Wkb, bk, Kb, 0);
    proj_gemm<<<pg, 128, 0, stream>>>(Xb, Wvb, bv, Vtb, 1);

    dim3 ag(BATCH * H_NUM, SEQ / 64);
    attn_kernel<<<ag, 128, 0, stream>>>(Qb, Kb, Vtb, bias, mask, out);
}
